// BasicTransformerBlock_35304631173827
// MI455X (gfx1250) — matrix operand order repaired and checked by reading, not compiled, not run on hardware
//
#include <hip/hip_runtime.h>
#include <math.h>

typedef __attribute__((ext_vector_type(16))) __bf16 v16bf;
typedef __attribute__((ext_vector_type(8)))  __bf16 v8bf;
typedef __attribute__((ext_vector_type(4)))  __bf16 v4bf;
typedef __attribute__((ext_vector_type(8)))  float  v8f;
typedef __attribute__((ext_vector_type(4)))  float  v4f;

// ---------- scalar fp32 -> bf16 (RNE) ----------

__device__ __forceinline__ __bf16 f2bf(float f) {
  unsigned u = __builtin_bit_cast(unsigned, f);
  unsigned r = (u + 0x7FFFu + ((u >> 16) & 1u)) >> 16;
  unsigned short h = (unsigned short)r;
  return __builtin_bit_cast(__bf16, h);
}

__device__ __forceinline__ v4bf f4_to_bf4(v4f f) {
#if __has_builtin(__builtin_amdgcn_cvt_pk_bf16_f32)
  auto lo = __builtin_amdgcn_cvt_pk_bf16_f32(f[0], f[1]);
  auto hi = __builtin_amdgcn_cvt_pk_bf16_f32(f[2], f[3]);
  unsigned u0 = __builtin_bit_cast(unsigned, lo);
  unsigned u1 = __builtin_bit_cast(unsigned, hi);
  unsigned long long uu = ((unsigned long long)u1 << 32) | u0;
  return __builtin_bit_cast(v4bf, uu);
#else
  v4bf r;
#pragma unroll
  for (int j = 0; j < 4; ++j) r[j] = f2bf(f[j]);
  return r;
#endif
}

// ---------- async global->LDS 16-byte copy (gfx1250 ASYNCcnt path) ----------
// Probe-confirmed signature: param 0 is AS(1) pointer to 'int __vector(4)'.

#if __has_builtin(__builtin_amdgcn_global_load_async_to_lds_b128) && __has_builtin(__builtin_amdgcn_s_wait_asynccnt)
#define USE_ASYNC_LDS 1
#else
#define USE_ASYNC_LDS 0
#endif

typedef int vsi4 __attribute__((vector_size(16)));

__device__ __forceinline__ void cp16_g2l(const __bf16* g, __bf16* l) {
#if USE_ASYNC_LDS
  __builtin_amdgcn_global_load_async_to_lds_b128(
      (__attribute__((address_space(1))) vsi4*)(g),
      (__attribute__((address_space(3))) vsi4*)(l), 0, 0);
#else
  *(v8bf*)l = *(const v8bf*)g;
#endif
}
__device__ __forceinline__ void cp_wait() {
#if USE_ASYNC_LDS
  __builtin_amdgcn_s_wait_asynccnt(0);
#endif
}

// ---------- 16-lane xor reductions via v_permlane16_b32 (VALU, no LDS) ----------

#if __has_builtin(__builtin_amdgcn_permlane16)
template <unsigned W0, unsigned W1>
__device__ __forceinline__ float permlane16f(float v) {
  unsigned i = __builtin_bit_cast(unsigned, v);
  unsigned r = __builtin_amdgcn_permlane16(i, i, W0, W1, false, false);
  return __builtin_bit_cast(float, r);
}
__device__ __forceinline__ float rmax16(float v) {
  v = fmaxf(v, permlane16f<0x67452301u, 0xEFCDAB89u>(v));  // xor 1
  v = fmaxf(v, permlane16f<0x54761032u, 0xDCFE98BAu>(v));  // xor 2
  v = fmaxf(v, permlane16f<0x32107654u, 0xBA98FEDCu>(v));  // xor 4
  v = fmaxf(v, permlane16f<0xFEDCBA98u, 0x76543210u>(v));  // xor 8
  return v;
}
__device__ __forceinline__ float rsum16(float v) {
  v += permlane16f<0x67452301u, 0xEFCDAB89u>(v);
  v += permlane16f<0x54761032u, 0xDCFE98BAu>(v);
  v += permlane16f<0x32107654u, 0xBA98FEDCu>(v);
  v += permlane16f<0xFEDCBA98u, 0x76543210u>(v);
  return v;
}
#else
__device__ __forceinline__ float rmax16(float v) {
  v = fmaxf(v, __shfl_xor(v, 1)); v = fmaxf(v, __shfl_xor(v, 2));
  v = fmaxf(v, __shfl_xor(v, 4)); v = fmaxf(v, __shfl_xor(v, 8));
  return v;
}
__device__ __forceinline__ float rsum16(float v) {
  v += __shfl_xor(v, 1); v += __shfl_xor(v, 2);
  v += __shfl_xor(v, 4); v += __shfl_xor(v, 8);
  return v;
}
#endif

__device__ __forceinline__ v8f wmma_bf16(v16bf a, v16bf b, v8f c) {
  return __builtin_amdgcn_wmma_f32_16x16x32_bf16(false, a, false, b, (short)0, c, false, false);
}

// A fragment: 16x32 tile from LDS row-major [m][k].
__device__ __forceinline__ v16bf load_a_frag(const __bf16* base, int stride, int m_off) {
  int lane = threadIdx.x & 31;
  int m = m_off + (lane & 15);
  int kb = (lane >> 4) * 16;
  const __bf16* row = base + m * stride;
  v8bf lo = *(const v8bf*)(row + kb);
  v8bf hi = *(const v8bf*)(row + kb + 8);
  v16bf r;
#pragma unroll
  for (int i = 0; i < 8; ++i) { r[i] = lo[i]; r[i + 8] = hi[i]; }
  return r;
}

// B fragment: 32x16 tile from LDS stored K-contiguous per column: base[n*stride + k].
__device__ __forceinline__ v16bf load_b_frag(const __bf16* base, int stride, int n_off) {
  int lane = threadIdx.x & 31;
  int n = n_off + (lane & 15);
  int kb = (lane >> 4) * 16;
  return *(const v16bf*)(base + n * stride + kb);
}

// ---------- weight prep: W (KxN fp32) -> WT (NxK bf16, k-contiguous) ----------

__global__ __launch_bounds__(256)
void wt_transpose_kernel(const float* __restrict__ W, __bf16* __restrict__ WT, int K, int N) {
  __shared__ float sT[64][65];
  int k0 = blockIdx.y * 64, n0 = blockIdx.x * 64;
  int tid = threadIdx.x;
#pragma unroll
  for (int i = 0; i < 4; ++i) {               // read 64x64 fp32 tile, coalesced
    int e4 = tid + i * 256;
    int r = e4 >> 4, c = (e4 & 15) * 4;
    v4f f = *(const v4f*)(W + (size_t)(k0 + r) * N + n0 + c);
#pragma unroll
    for (int j = 0; j < 4; ++j) sT[r][c + j] = f[j];
  }
  __syncthreads();
#pragma unroll
  for (int i = 0; i < 4; ++i) {               // write transposed bf16, coalesced
    int e4 = tid + i * 256;
    int n = e4 >> 4, k4 = (e4 & 15) * 4;
    v4f f;
#pragma unroll
    for (int j = 0; j < 4; ++j) f[j] = sT[k4 + j][n];
    *(v4bf*)(WT + (size_t)(n0 + n) * K + k0 + k4) = f4_to_bf4(f);
  }
}

// ---------- tiny GEMM for t @ W + b  (M = 4 rows only) ----------

__global__ void emb_kernel(const float* __restrict__ t, const float* __restrict__ W,
                           const float* __restrict__ bias, float* __restrict__ out,
                           int D, int N2) {
  int n = blockIdx.x * blockDim.x + threadIdx.x;
  int b = blockIdx.y;
  if (n >= N2) return;
  const float* tr = t + (size_t)b * D;
  float s = bias[n];
  for (int k = 0; k < D; ++k) s += tr[k] * W[(size_t)k * N2 + n];
  out[(size_t)b * N2 + n] = s;
}

// ---------- AdaLayerNorm, D=1024, 256 threads: one float4 per thread ----------

__global__ __launch_bounds__(256)
void ada_ln_kernel(const float* __restrict__ x, const float* __restrict__ emb,
                   float* __restrict__ h, int rows_per_batch) {
  const int D = 1024;
  int row = blockIdx.x;
  int b = row / rows_per_batch;
  const float* xr = x + (size_t)row * D;
  int d4 = threadIdx.x * 4;
  v4f v = *(const v4f*)(xr + d4);
  float s = v[0] + v[1] + v[2] + v[3];
  float s2 = v[0] * v[0] + v[1] * v[1] + v[2] * v[2] + v[3] * v[3];
#pragma unroll
  for (int off = 16; off > 0; off >>= 1) { s += __shfl_xor(s, off); s2 += __shfl_xor(s2, off); }
  __shared__ float rs[8], rs2[8];
  int wid = threadIdx.x >> 5, lane = threadIdx.x & 31;
  if (lane == 0) { rs[wid] = s; rs2[wid] = s2; }
  __syncthreads();
  if (threadIdx.x == 0) {
    float a = 0.f, a2 = 0.f;
#pragma unroll
    for (int i = 0; i < 8; ++i) { a += rs[i]; a2 += rs2[i]; }
    rs[0] = a; rs2[0] = a2;
  }
  __syncthreads();
  float mu = rs[0] * (1.f / D);
  float var = rs2[0] * (1.f / D) - mu * mu;
  float rinv = rsqrtf(var + 1e-5f);
  const float* eb = emb + (size_t)b * 2 * D;
  v4f sc = *(const v4f*)(eb + d4);
  v4f sh = *(const v4f*)(eb + D + d4);
  v4f o;
#pragma unroll
  for (int j = 0; j < 4; ++j) o[j] = (v[j] - mu) * rinv * (1.f + sc[j]) + sh[j];
  *(v4f*)(h + (size_t)row * D + d4) = o;
}

// ---------- bf16-WMMA GEMM: C = A(MxK,f32) @ BT(NxK,bf16)^T [+bias] [+resid] ----------
// Macro tile 128x128, 8 waves as 4x2, wave tile 32x64. B tiles staged via async-to-LDS.

template <int N, int K, bool HAS_BIAS, bool HAS_RES, bool BF16_OUT>
__global__ __launch_bounds__(256)
void gemm_bf16_kernel(const float* __restrict__ A, const __bf16* __restrict__ BT,
                      const float* __restrict__ bias, const float* resid,
                      void* Cout, float oscale) {
  __shared__ __attribute__((aligned(32))) __bf16 sA[128 * 32];   // [m][k]
  __shared__ __attribute__((aligned(32))) __bf16 sBT[128 * 32];  // [n][k]
  int m0 = blockIdx.y * 128;
  int n0 = blockIdx.x * 128;
  int tid = threadIdx.x;
  int wave = tid >> 5;
  int wm = (wave >> 1) * 32;
  int wn = (wave & 1) * 64;
  v8f acc[2][4] = {};
  for (int k0 = 0; k0 < K; k0 += 32) {
    __syncthreads();
    const __bf16* Bt = BT + (size_t)n0 * K + k0;
#pragma unroll
    for (int i = 0; i < 2; ++i) {              // 128x32 bf16 B tile: pure 16B copies
      int u = tid + i * 256;
      int n = u >> 1, half = (u & 1) * 16;
      cp16_g2l(Bt + (size_t)n * K + half, sBT + n * 32 + half);
    }
#pragma unroll
    for (int i = 0; i < 4; ++i) {              // 128x32 A tile: fp32 -> bf16
      int e4 = tid + i * 256;
      int r = e4 >> 3, c = (e4 & 7) * 4;
      v4f f = *(const v4f*)(A + (size_t)(m0 + r) * K + k0 + c);
      *(v4bf*)(sA + r * 32 + c) = f4_to_bf4(f);
    }
    cp_wait();
    __syncthreads();
    v16bf a0 = load_a_frag(sA, 32, wm);
    v16bf a1 = load_a_frag(sA, 32, wm + 16);
    v16bf b0 = load_b_frag(sBT, 32, wn);
    v16bf b1 = load_b_frag(sBT, 32, wn + 16);
    v16bf b2 = load_b_frag(sBT, 32, wn + 32);
    v16bf b3 = load_b_frag(sBT, 32, wn + 48);
    acc[0][0] = wmma_bf16(a0, b0, acc[0][0]);
    acc[0][1] = wmma_bf16(a0, b1, acc[0][1]);
    acc[0][2] = wmma_bf16(a0, b2, acc[0][2]);
    acc[0][3] = wmma_bf16(a0, b3, acc[0][3]);
    acc[1][0] = wmma_bf16(a1, b0, acc[1][0]);
    acc[1][1] = wmma_bf16(a1, b1, acc[1][1]);
    acc[1][2] = wmma_bf16(a1, b2, acc[1][2]);
    acc[1][3] = wmma_bf16(a1, b3, acc[1][3]);
  }
  int lane = tid & 31;
  int ncol = lane & 15;
  int mrow8 = (lane >> 4) * 8;
  int gm0 = m0 + wm + mrow8;
  int gn0 = n0 + wn + ncol;
  if constexpr (BF16_OUT) {
    __bf16* Cb = (__bf16*)Cout + (size_t)gm0 * N + gn0;
#pragma unroll
    for (int r = 0; r < 2; ++r)
#pragma unroll
      for (int c = 0; c < 4; ++c)
#pragma unroll
        for (int vr = 0; vr < 8; ++vr)
          Cb[(size_t)(r * 16 + vr) * N + c * 16] = f2bf(acc[r][c][vr] * oscale);
  } else {
    float* Cp = (float*)Cout + (size_t)gm0 * N + gn0;
    const float* Rp = HAS_RES ? (resid + (size_t)gm0 * N + gn0) : nullptr;
    float bv[4];
#pragma unroll
    for (int c = 0; c < 4; ++c) bv[c] = HAS_BIAS ? bias[gn0 + c * 16] : 0.f;
#pragma unroll
    for (int r = 0; r < 2; ++r)
#pragma unroll
      for (int c = 0; c < 4; ++c)
#pragma unroll
        for (int vr = 0; vr < 8; ++vr) {
          size_t off = (size_t)(r * 16 + vr) * N + c * 16;
          float v = acc[r][c][vr] + bv[c];
          if (HAS_RES) v += Rp[off];
          Cp[off] = v;
        }
  }
}

// ---------- flash attention over bf16 Q/K/V (Q pre-scaled), fp32 O ----------

__global__ __launch_bounds__(128)
void flash_attn_kernel(const __bf16* __restrict__ Q, const __bf16* __restrict__ K,
                       const __bf16* __restrict__ V, float* __restrict__ O,
                       int qlen, int kvlen) {
  const int STR = 1024, DH = 64;
  __shared__ __attribute__((aligned(32))) __bf16 sQ[64 * 64];     // [m][d]
  __shared__ __attribute__((aligned(32))) __bf16 sK[32 * 64];     // [j][d]
  __shared__ __attribute__((aligned(32))) __bf16 sV[64 * 32];     // [d][j]
  __shared__ __attribute__((aligned(32))) __bf16 sP[4 * 16 * 32]; // per-wave P [m][j]
  int b = blockIdx.z, hh = blockIdx.y;
  int q0 = blockIdx.x * 64;
  int tid = threadIdx.x, wid = tid >> 5, lane = tid & 31;
  const __bf16* Qb = Q + ((size_t)(b * qlen + q0)) * STR + hh * DH;
  const __bf16* Kb = K + ((size_t)b * kvlen) * STR + hh * DH;
  const __bf16* Vb = V + ((size_t)b * kvlen) * STR + hh * DH;
#pragma unroll
  for (int i = 0; i < 4; ++i) {               // 64x64 bf16 Q tile: 16B copies
    int u = tid + i * 128;
    int r = u >> 3, off = (u & 7) * 8;
    cp16_g2l(Qb + (size_t)r * STR + off, sQ + r * 64 + off);
  }
  int m_base = wid * 16;
  v8f o0 = {}, o1 = {}, o2 = {}, o3 = {};
  float m_r[8], l_r[8];
#pragma unroll
  for (int r = 0; r < 8; ++r) { m_r[r] = -1e30f; l_r[r] = 0.f; }
  cp_wait();
  __syncthreads();
  int mrow8 = (lane >> 4) * 8;
  int ncol = lane & 15;
  for (int j0 = 0; j0 < kvlen; j0 += 32) {
#pragma unroll
    for (int i = 0; i < 2; ++i) {             // 32x64 K tile: 16B copies
      int u = tid + i * 128;
      int r = u >> 3, off = (u & 7) * 8;
      cp16_g2l(Kb + (size_t)(j0 + r) * STR + off, sK + r * 64 + off);
    }
#pragma unroll
    for (int i = 0; i < 2; ++i) {             // 32x64 V tile, transposed into [d][j]
      int u = tid + i * 128;
      int r = u >> 3, off = (u & 7) * 8;
      v8bf fv = *(const v8bf*)(Vb + (size_t)(j0 + r) * STR + off);
#pragma unroll
      for (int j = 0; j < 8; ++j) sV[(off + j) * 32 + r] = fv[j];
    }
    cp_wait();
    __syncthreads();
    // S(16x32) = Q_w(16x64) @ K^T(64x32)
    v8f s0 = {}, s1 = {};
#pragma unroll
    for (int d0 = 0; d0 < 64; d0 += 32) {
      v16bf aq  = load_a_frag(sQ + d0, 64, m_base);
      v16bf bk0 = load_b_frag(sK + d0, 64, 0);
      v16bf bk1 = load_b_frag(sK + d0, 64, 16);
      s0 = wmma_bf16(aq, bk0, s0);
      s1 = wmma_bf16(aq, bk1, s1);
    }
    // online softmax (all-VALU permlane16 reductions)
    __bf16* sPw = sP + wid * (16 * 32);
#pragma unroll
    for (int r = 0; r < 8; ++r) {
      float mx = rmax16(fmaxf(s0[r], s1[r]));
      float mnew = fmaxf(m_r[r], mx);
      float corr = __expf(m_r[r] - mnew);
      float p0 = __expf(s0[r] - mnew);
      float p1 = __expf(s1[r] - mnew);
      float rsum = rsum16(p0 + p1);
      l_r[r] = l_r[r] * corr + rsum;
      m_r[r] = mnew;
      o0[r] *= corr; o1[r] *= corr; o2[r] *= corr; o3[r] *= corr;
      sPw[(mrow8 + r) * 32 + ncol]      = f2bf(p0);
      sPw[(mrow8 + r) * 32 + 16 + ncol] = f2bf(p1);
    }
    // O(16x64) += P(16x32) @ V(32x64)
    v16bf ap  = load_a_frag(sPw, 32, 0);
    v16bf bv0 = load_b_frag(sV, 32, 0);
    v16bf bv1 = load_b_frag(sV, 32, 16);
    v16bf bv2 = load_b_frag(sV, 32, 32);
    v16bf bv3 = load_b_frag(sV, 32, 48);
    o0 = wmma_bf16(ap, bv0, o0);
    o1 = wmma_bf16(ap, bv1, o1);
    o2 = wmma_bf16(ap, bv2, o2);
    o3 = wmma_bf16(ap, bv3, o3);
    __syncthreads();
  }
#pragma unroll
  for (int r = 0; r < 8; ++r) {
    float inv = 1.0f / l_r[r];
    size_t row = (size_t)(b * qlen + q0 + m_base + mrow8 + r) * STR + hh * DH;
    O[row + 0  + ncol] = o0[r] * inv;
    O[row + 16 + ncol] = o1[r] * inv;
    O[row + 32 + ncol] = o2[r] * inv;
    O[row + 48 + ncol] = o3[r] * inv;
  }
}

// ---------- GEGLU: g = a * gelu(gate), exact erf gelu, float4 ----------

__global__ __launch_bounds__(256)
void geglu_kernel(const float* __restrict__ p, float* __restrict__ g, int rows) {
  const int FF = 4096;
  int i = blockIdx.x * 256 + threadIdx.x;
  int r = i / (FF / 4);
  int f4 = (i % (FF / 4)) * 4;
  if (r >= rows) return;
  v4f a  = *(const v4f*)(p + (size_t)r * (2 * FF) + f4);
  v4f xg = *(const v4f*)(p + (size_t)r * (2 * FF) + FF + f4);
  v4f o;
#pragma unroll
  for (int j = 0; j < 4; ++j) {
    float gelu = 0.5f * xg[j] * (1.0f + erff(xg[j] * 0.7071067811865475f));
    o[j] = a[j] * gelu;
  }
  *(v4f*)(g + (size_t)r * FF + f4) = o;
}

// ---------- launcher ----------

extern "C" void kernel_launch(void* const* d_in, const int* in_sizes, int n_in,
                              void* d_out, int out_size, void* d_ws, size_t ws_size,
                              hipStream_t stream) {
  (void)in_sizes; (void)n_in; (void)out_size; (void)ws_size;
  const float* x    = (const float*)d_in[0];
  const float* t    = (const float*)d_in[1];
  const float* ctx  = (const float*)d_in[2];
  const float* n1_w = (const float*)d_in[3];
  const float* n1_b = (const float*)d_in[4];
  const float* n2_w = (const float*)d_in[5];
  const float* n2_b = (const float*)d_in[6];
  const float* n3_w = (const float*)d_in[7];
  const float* n3_b = (const float*)d_in[8];
  const float* q1   = (const float*)d_in[9];
  const float* k1   = (const float*)d_in[10];
  const float* v1   = (const float*)d_in[11];
  const float* o1w  = (const float*)d_in[12];
  const float* o1b  = (const float*)d_in[13];
  const float* q2   = (const float*)d_in[14];
  const float* k2   = (const float*)d_in[15];
  const float* v2   = (const float*)d_in[16];
  const float* o2w  = (const float*)d_in[17];
  const float* o2b  = (const float*)d_in[18];
  const float* ffw1 = (const float*)d_in[19];
  const float* ffb1 = (const float*)d_in[20];
  const float* ffw2 = (const float*)d_in[21];
  const float* ffb2 = (const float*)d_in[22];
  float* out = (float*)d_out;

  const int Bn = 4, Nn = 2048, Dn = 1024, Jn = 256, Hn = 16;
  const int ROWS = Bn * Nn;             // 8192
  const int CROWS = Bn * Jn;            // 1024
  const size_t R = (size_t)ROWS * Dn;   // 8M floats per activation buffer

  float* ws   = (float*)d_ws;
  float* xcur = ws;                     // fp32 residual stream
  float* hbuf = ws + R;                 // fp32 AdaLN output
  float* bufA = ws + 2 * R;             // bf16 Q   / fp32 p_chunk
  float* bufB = ws + 3 * R;             // bf16 K   / fp32 g_chunk
  float* bufC = ws + 4 * R;             // bf16 V
  float* bufD = ws + 5 * R;             // fp32 attention output
  float* emb  = ws + 6 * R;             // 3 x (4 x 2048), padded to 32768 floats

  // pre-transposed bf16 weights [N][K]
  __bf16* wt  = (__bf16*)(ws + 6 * R + 32768);
  __bf16* q1T = wt;
  __bf16* k1T = q1T + (size_t)1024 * 1024;
  __bf16* v1T = k1T + (size_t)1024 * 1024;
  __bf16* o1T = v1T + (size_t)1024 * 1024;
  __bf16* q2T = o1T + (size_t)1024 * 1024;
  __bf16* k2T = q2T + (size_t)1024 * 1024;
  __bf16* v2T = k2T + (size_t)1024 * 768;
  __bf16* o2T = v2T + (size_t)1024 * 768;
  __bf16* f1T = o2T + (size_t)1024 * 1024;
  __bf16* f2T = f1T + (size_t)8192 * 1024;

  dim3 blk256(256), blk128(128);

  // ---- one-time per launch: weight transpose+convert ----
  wt_transpose_kernel<<<dim3(16, 16),  blk256, 0, stream>>>(q1,   q1T, 1024, 1024);
  wt_transpose_kernel<<<dim3(16, 16),  blk256, 0, stream>>>(k1,   k1T, 1024, 1024);
  wt_transpose_kernel<<<dim3(16, 16),  blk256, 0, stream>>>(v1,   v1T, 1024, 1024);
  wt_transpose_kernel<<<dim3(16, 16),  blk256, 0, stream>>>(o1w,  o1T, 1024, 1024);
  wt_transpose_kernel<<<dim3(16, 16),  blk256, 0, stream>>>(q2,   q2T, 1024, 1024);
  wt_transpose_kernel<<<dim3(16, 12),  blk256, 0, stream>>>(k2,   k2T, 768,  1024);
  wt_transpose_kernel<<<dim3(16, 12),  blk256, 0, stream>>>(v2,   v2T, 768,  1024);
  wt_transpose_kernel<<<dim3(16, 16),  blk256, 0, stream>>>(o2w,  o2T, 1024, 1024);
  wt_transpose_kernel<<<dim3(128, 16), blk256, 0, stream>>>(ffw1, f1T, 1024, 8192);
  wt_transpose_kernel<<<dim3(16, 64),  blk256, 0, stream>>>(ffw2, f2T, 4096, 1024);

  // time embeddings
  emb_kernel<<<dim3(8, Bn), blk256, 0, stream>>>(t, n1_w, n1_b, emb,                   Dn, 2 * Dn);
  emb_kernel<<<dim3(8, Bn), blk256, 0, stream>>>(t, n2_w, n2_b, emb + Bn * 2 * Dn,     Dn, 2 * Dn);
  emb_kernel<<<dim3(8, Bn), blk256, 0, stream>>>(t, n3_w, n3_b, emb + 2 * Bn * 2 * Dn, Dn, 2 * Dn);

  // ---- block 1: self attention ----
  ada_ln_kernel<<<ROWS, blk256, 0, stream>>>(x, emb, hbuf, Nn);
  gemm_bf16_kernel<1024, 1024, false, false, true><<<dim3(8, ROWS / 128), blk256, 0, stream>>>(hbuf, q1T, nullptr, nullptr, bufA, 0.125f);
  gemm_bf16_kernel<1024, 1024, false, false, true><<<dim3(8, ROWS / 128), blk256, 0, stream>>>(hbuf, k1T, nullptr, nullptr, bufB, 1.0f);
  gemm_bf16_kernel<1024, 1024, false, false, true><<<dim3(8, ROWS / 128), blk256, 0, stream>>>(hbuf, v1T, nullptr, nullptr, bufC, 1.0f);
  flash_attn_kernel<<<dim3(Nn / 64, Hn, Bn), blk128, 0, stream>>>(
      (const __bf16*)bufA, (const __bf16*)bufB, (const __bf16*)bufC, bufD, Nn, Nn);
  gemm_bf16_kernel<1024, 1024, true, true, false><<<dim3(8, ROWS / 128), blk256, 0, stream>>>(bufD, o1T, o1b, x, xcur, 1.0f);

  // ---- block 2: cross attention ----
  ada_ln_kernel<<<ROWS, blk256, 0, stream>>>(xcur, emb + Bn * 2 * Dn, hbuf, Nn);
  gemm_bf16_kernel<1024, 1024, false, false, true><<<dim3(8, ROWS / 128), blk256, 0, stream>>>(hbuf, q2T, nullptr, nullptr, bufA, 0.125f);
  gemm_bf16_kernel<1024, 768, false, false, true><<<dim3(8, CROWS / 128), blk256, 0, stream>>>(ctx, k2T, nullptr, nullptr, bufB, 1.0f);
  gemm_bf16_kernel<1024, 768, false, false, true><<<dim3(8, CROWS / 128), blk256, 0, stream>>>(ctx, v2T, nullptr, nullptr, bufC, 1.0f);
  flash_attn_kernel<<<dim3(Nn / 64, Hn, Bn), blk128, 0, stream>>>(
      (const __bf16*)bufA, (const __bf16*)bufB, (const __bf16*)bufC, bufD, Nn, Jn);
  gemm_bf16_kernel<1024, 1024, true, true, false><<<dim3(8, ROWS / 128), blk256, 0, stream>>>(bufD, o2T, o2b, xcur, xcur, 1.0f);

  // ---- block 3: GEGLU feed-forward (chunked over rows) ----
  ada_ln_kernel<<<ROWS, blk256, 0, stream>>>(xcur, emb + 2 * Bn * 2 * Dn, hbuf, Nn);
  const int CH = 1024;
  for (int c0 = 0; c0 < ROWS; c0 += CH) {
    gemm_bf16_kernel<8192, 1024, true, false, false><<<dim3(64, CH / 128), blk256, 0, stream>>>(
        hbuf + (size_t)c0 * Dn, f1T, ffb1, nullptr, bufA, 1.0f);
    geglu_kernel<<<(CH * 4096) / (4 * 256), blk256, 0, stream>>>(bufA, bufB, CH);
    gemm_bf16_kernel<1024, 4096, true, true, false><<<dim3(8, CH / 128), blk256, 0, stream>>>(
        bufB, f2T, ffb2, xcur + (size_t)c0 * Dn, out + (size_t)c0 * Dn, 1.0f);
  }
}
